// Net_47493748359894
// MI455X (gfx1250) — matrix operand order repaired and checked by reading, not compiled, not run on hardware
//
#include <hip/hip_runtime.h>
#include <hip/hip_bf16.h>

typedef __attribute__((ext_vector_type(16))) __bf16 v16bf;
typedef __attribute__((ext_vector_type(8)))  float  v8f;

// ---------- helpers ----------
__device__ __forceinline__ unsigned bfbits(float f) {
    unsigned u = __builtin_bit_cast(unsigned, f);
    unsigned r = u + 0x7FFFu + ((u >> 16) & 1u);      // round-to-nearest-even
    return r >> 16;
}
__device__ __forceinline__ unsigned pack2(float lo, float hi) {
    return (bfbits(hi) << 16) | bfbits(lo);
}
// order-preserving float <-> uint for atomic max
__device__ __forceinline__ unsigned encf(float f) {
    unsigned u = __builtin_bit_cast(unsigned, f);
    return (u & 0x80000000u) ? ~u : (u | 0x80000000u);
}
__device__ __forceinline__ float decf(unsigned u) {
    return (u & 0x80000000u) ? __builtin_bit_cast(float, u & 0x7FFFFFFFu)
                             : __builtin_bit_cast(float, ~u);
}

// ---------- pack A (M x K, row major fp32) into WMMA A-fragment order ----------
// Fragment f = (mtile*nk + ks): 32 lanes x 16 bf16 (32 B/lane, contiguous).
// lane half h: elems 0..7  -> K = ks*32 + 8h + e
//              elems 8..15 -> K = ks*32 + 16 + 8h + (e-8)
__global__ void pack_a(const float* __restrict__ A, __bf16* __restrict__ Apk,
                       int M, int K)
{
    int nk = K >> 5;
    long long t = (long long)blockIdx.x * blockDim.x + threadIdx.x;
    long long total = (long long)(M >> 4) * nk * 32;
    if (t >= total) return;
    int lane = (int)(t & 31);
    long long q = t >> 5;
    int ks = (int)(q % nk);
    long long mt = q / nk;
    int half = lane >> 4, l16 = lane & 15;

    const float* p = A + (mt * 16 + l16) * (long long)K + ks * 32 + half * 8;
    unsigned w0 = pack2(p[0],  p[1]),  w1 = pack2(p[2],  p[3]);
    unsigned w2 = pack2(p[4],  p[5]),  w3 = pack2(p[6],  p[7]);
    unsigned w4 = pack2(p[16], p[17]), w5 = pack2(p[18], p[19]);
    unsigned w6 = pack2(p[20], p[21]), w7 = pack2(p[22], p[23]);
    uint4* o = (uint4*)(Apk + t * 16);
    o[0] = make_uint4(w0, w1, w2, w3);
    o[1] = make_uint4(w4, w5, w6, w7);
}

// ---------- pack B (K x Nd, row major fp32) into WMMA B-fragment order ----------
// Fragment f = (ntile*nk + ks): lane elems e<8 -> K = ks*32 + 8*half + e, e>=8 -> K = ks*32 + 16 + 8*half + (e-8), N = ntile*16 + l16
__global__ void pack_b(const float* __restrict__ B, __bf16* __restrict__ Bpk,
                       int K, int Nd)
{
    int nk = K >> 5;
    long long t = (long long)blockIdx.x * blockDim.x + threadIdx.x;
    long long total = (long long)(Nd >> 4) * nk * 32;
    if (t >= total) return;
    int lane = (int)(t & 31);
    long long q = t >> 5;
    int ks = (int)(q % nk);
    int nt = (int)(q / nk);
    int half = lane >> 4, l16 = lane & 15;

    const float* p = B + (long long)(ks * 32 + half * 8) * Nd + nt * 16 + l16;
    unsigned w[8];
#pragma unroll
    for (int j = 0; j < 8; ++j)
        w[j] = pack2(p[(long long)(2 * j + (j >> 2) * 8) * Nd], p[(long long)(2 * j + 1 + (j >> 2) * 8) * Nd]);
    uint4* o = (uint4*)(Bpk + t * 16);
    o[0] = make_uint4(w[0], w[1], w[2], w[3]);
    o[1] = make_uint4(w[4], w[5], w[6], w[7]);
}

// ---------- bf16 WMMA GEMM on pre-packed fragments ----------
// Each wave computes a 32x32 output tile (2x2 of 16x16), 4 WMMAs per K-step.
__global__ __launch_bounds__(256) void gemm_bf16_wmma(
    const __bf16* __restrict__ Apk, const __bf16* __restrict__ Bpk,
    float* __restrict__ C, int M, int Nd, int K)
{
    const int nk  = K >> 5;
    const int nt2 = Nd >> 5;                     // 32-wide supertiles in N
    const int mt2 = M >> 5;                      // 32-tall supertiles in M
    const int wid = blockIdx.x * (blockDim.x >> 5) + (threadIdx.x >> 5);
    if (wid >= nt2 * mt2) return;
    const int wm = wid / nt2, wn = wid % nt2;
    const int lane = threadIdx.x & 31;
    const int half = lane >> 4, l16 = lane & 15;

    const v16bf* Af = (const v16bf*)Apk;
    const v16bf* Bf = (const v16bf*)Bpk;
    const long long a0base = ((long long)(wm * 2 + 0) * nk) * 32 + lane;
    const long long a1base = ((long long)(wm * 2 + 1) * nk) * 32 + lane;
    const long long b0base = ((long long)(wn * 2 + 0) * nk) * 32 + lane;
    const long long b1base = ((long long)(wn * 2 + 1) * nk) * 32 + lane;

    v8f acc00 = {}, acc01 = {}, acc10 = {}, acc11 = {};
    for (int ks = 0; ks < nk; ++ks) {
        const long long o = (long long)ks * 32;
        v16bf A0 = Af[a0base + o];
        v16bf A1 = Af[a1base + o];
        v16bf B0 = Bf[b0base + o];
        v16bf B1 = Bf[b1base + o];
        acc00 = __builtin_amdgcn_wmma_f32_16x16x32_bf16(false, A0, false, B0, (short)0, acc00, false, false);
        acc01 = __builtin_amdgcn_wmma_f32_16x16x32_bf16(false, A0, false, B1, (short)0, acc01, false, false);
        acc10 = __builtin_amdgcn_wmma_f32_16x16x32_bf16(false, A1, false, B0, (short)0, acc10, false, false);
        acc11 = __builtin_amdgcn_wmma_f32_16x16x32_bf16(false, A1, false, B1, (short)0, acc11, false, false);
    }

    // C/D layout: vgpr v, lane -> row = tile_m + v + 8*half, col = tile_n + l16
    const int rm = wm * 32, rn = wn * 32;
    float* C00 = C + (long long)(rm + half * 8) * Nd + rn + l16;
    float* C10 = C + (long long)(rm + 16 + half * 8) * Nd + rn + l16;
#pragma unroll
    for (int v = 0; v < 8; ++v) {
        C00[(long long)v * Nd]      = acc00[v];
        C00[(long long)v * Nd + 16] = acc01[v];
        C10[(long long)v * Nd]      = acc10[v];
        C10[(long long)v * Nd + 16] = acc11[v];
    }
}

// ---------- attention scores: el/er[n*4+h] = sum_f h[n,h,f]*attn[h,f] ----------
__global__ void attn_scores(const float* __restrict__ h,
                            const float* __restrict__ al,
                            const float* __restrict__ ar,
                            float* __restrict__ el, float* __restrict__ er, int N)
{
    int idx = blockIdx.x * blockDim.x + threadIdx.x;   // (n*4 + head)
    if (idx >= N * 4) return;
    int hd = idx & 3;
    const float* hp  = h  + (long long)idx * 128;      // (n*4+hd)*128 == n*512 + hd*128
    const float* alp = al + hd * 128;
    const float* arp = ar + hd * 128;
    float sl = 0.f, sr = 0.f;
#pragma unroll 4
    for (int f = 0; f < 128; ++f) { float v = hp[f]; sl += v * alp[f]; sr += v * arp[f]; }
    el[idx] = sl; er[idx] = sr;
}

// ---------- init emax/denom and zero the accumulation buffer ----------
__global__ void init_buffers(unsigned* __restrict__ emax_u, float* __restrict__ denom,
                             float* __restrict__ acc, int n4, long long nf)
{
    long long i = (long long)blockIdx.x * blockDim.x + threadIdx.x;
    if (i < n4) { emax_u[i] = 0x007FFFFFu; /* enc(-inf) */ denom[i] = 0.f; }
    if (i < nf) acc[i] = 0.f;
}

// ---------- edge pass 1: leaky_relu + segment max ----------
__global__ void edge_max(const int* __restrict__ src, const int* __restrict__ dst,
                         const float* __restrict__ el, const float* __restrict__ er,
                         float* __restrict__ eedge, unsigned* __restrict__ emax_u, int E)
{
    int i = blockIdx.x * blockDim.x + threadIdx.x;     // edge*4 + head
    if (i >= E * 4) return;
    int e = i >> 2, hd = i & 3;
    int s = src[e], d = dst[e];
    float v = el[s * 4 + hd] + er[d * 4 + hd];
    v = (v > 0.f) ? v : 0.2f * v;                      // LeakyReLU(0.2)
    eedge[i] = v;
    atomicMax(&emax_u[d * 4 + hd], encf(v));
}

// ---------- edge pass 2: exp(e - max) + segment sum ----------
__global__ void edge_expsum(const int* __restrict__ dst, float* __restrict__ eedge,
                            const unsigned* __restrict__ emax_u,
                            float* __restrict__ denom, int E)
{
    int i = blockIdx.x * blockDim.x + threadIdx.x;
    if (i >= E * 4) return;
    int e = i >> 2, hd = i & 3;
    int d = dst[e];
    float ee = __expf(eedge[i] - decf(emax_u[d * 4 + hd]));
    eedge[i] = ee;
    atomicAdd(&denom[d * 4 + hd], ee);
}

// ---------- edge pass 3: alpha-weighted scatter-add (h is L2-resident) ----------
__global__ __launch_bounds__(256) void edge_scatter(
    const int* __restrict__ src, const int* __restrict__ dst,
    const float* __restrict__ h, const float* __restrict__ eedge,
    const float* __restrict__ denom, float* __restrict__ acc, int E)
{
    int e = blockIdx.x;
    if (e >= E) return;
    int s = src[e], d = dst[e];
    int t = threadIdx.x;
#pragma unroll
    for (int r = 0; r < 2; ++r) {
        int c  = t + r * 256;                          // 0..511
        int hd = c >> 7;
        float alpha = eedge[e * 4 + hd] / denom[d * 4 + hd];
        atomicAdd(&acc[(long long)d * 512 + c], h[(long long)s * 512 + c] * alpha);
    }
}

// ---------- x = elu(x + bias) ----------
__global__ void bias_elu(float* __restrict__ x, const float* __restrict__ bias,
                         long long total, int width)
{
    long long i = (long long)blockIdx.x * blockDim.x + threadIdx.x;
    if (i >= total) return;
    float v = x[i] + bias[i % width];
    x[i] = (v > 0.f) ? v : (__expf(v) - 1.0f);
}

extern "C" void kernel_launch(void* const* d_in, const int* in_sizes, int n_in,
                              void* d_out, int out_size, void* d_ws, size_t ws_size,
                              hipStream_t stream) {
    const float* feature = (const float*)d_in[0];
    const float* W1  = (const float*)d_in[1];
    const float* al1 = (const float*)d_in[2];
    const float* ar1 = (const float*)d_in[3];
    const float* b1  = (const float*)d_in[4];
    const float* W2  = (const float*)d_in[5];
    const float* al2 = (const float*)d_in[6];
    const float* ar2 = (const float*)d_in[7];
    const float* b2  = (const float*)d_in[8];
    const float* Wfc = (const float*)d_in[9];
    const float* bfc = (const float*)d_in[10];
    const int*   src = (const int*)d_in[11];
    const int*   dst = (const int*)d_in[12];
    float* out = (float*)d_out;

    const int IN_F = 1280, HID = 512;
    const int N = in_sizes[0] / IN_F;
    const int E = in_sizes[11];
    const long long NF = (long long)N * HID;

    // -------- workspace layout (lifetimes overlapped) --------
    float* ws      = (float*)d_ws;
    float* bufA    = ws;                                  // gemm output h         (N*512 f32)
    float* bufB    = bufA + NF;                           // L1 scatter / gemm2 in (N*512 f32)
    float* region2 = bufB + NF;                           // N*640 f32:
    float* bufC    = region2;                             //   L2 scatter target   (N*512 f32)
    __bf16* Apk1   = (__bf16*)region2;                    //   packed feature      (N*1280 bf16) [dead before bufC used]
    __bf16* Apk2   = (__bf16*)(region2 + (long long)N * 640);  // packed 512-wide A (N*512 bf16)
    float* smallp  = region2 + (long long)N * 640 + (long long)N * 256;
    float*    el     = smallp;                            // (N*4)
    float*    er     = el + (long long)N * 4;             // (N*4)
    unsigned* emax_u = (unsigned*)(er + (long long)N * 4);// (N*4)
    float*    denom  = (float*)(emax_u + (long long)N * 4);// (N*4)
    float*    eedge  = denom + (long long)N * 4;          // (E*4)
    __bf16*   Bpk    = (__bf16*)(eedge + (long long)E * 4); // packed B (<= 1280*512 bf16)

    const int thr = 256;
    int gN4 = (N * 4 + thr - 1) / thr;
    int gE4 = (E * 4 + thr - 1) / thr;
    int gNF = (int)((NF + thr - 1) / thr);
    auto gemm_blocks = [&](int M, int Nd) {
        long long waves = (long long)(M >> 5) * (Nd >> 5);
        return (int)((waves + 7) / 8);
    };
    auto pack_blocks = [&](long long rows16, int K) {
        long long t = rows16 * (K >> 5) * 32;
        return (int)((t + thr - 1) / thr);
    };

    // ---------------- layer 1 ----------------
    pack_b<<<pack_blocks(HID >> 4, IN_F), thr, 0, stream>>>(W1, Bpk, IN_F, HID);
    pack_a<<<pack_blocks(N >> 4, IN_F), thr, 0, stream>>>(feature, Apk1, N, IN_F);
    gemm_bf16_wmma<<<gemm_blocks(N, HID), thr, 0, stream>>>(Apk1, Bpk, bufA, N, HID, IN_F);
    attn_scores<<<gN4, thr, 0, stream>>>(bufA, al1, ar1, el, er, N);
    init_buffers<<<gNF, thr, 0, stream>>>(emax_u, denom, bufB, N * 4, NF);
    edge_max<<<gE4, thr, 0, stream>>>(src, dst, el, er, eedge, emax_u, E);
    edge_expsum<<<gE4, thr, 0, stream>>>(dst, eedge, emax_u, denom, E);
    edge_scatter<<<E, 256, 0, stream>>>(src, dst, bufA, eedge, denom, bufB, E);
    bias_elu<<<gNF, thr, 0, stream>>>(bufB, b1, NF, HID);

    // ---------------- layer 2 ----------------
    pack_b<<<pack_blocks(HID >> 4, HID), thr, 0, stream>>>(W2, Bpk, HID, HID);
    pack_a<<<pack_blocks(N >> 4, HID), thr, 0, stream>>>(bufB, Apk2, N, HID);
    gemm_bf16_wmma<<<gemm_blocks(N, HID), thr, 0, stream>>>(Apk2, Bpk, bufA, N, HID, HID);
    attn_scores<<<gN4, thr, 0, stream>>>(bufA, al2, ar2, el, er, N);
    init_buffers<<<gNF, thr, 0, stream>>>(emax_u, denom, bufC, N * 4, NF);  // Apk1 dead now
    edge_max<<<gE4, thr, 0, stream>>>(src, dst, el, er, eedge, emax_u, E);
    edge_expsum<<<gE4, thr, 0, stream>>>(dst, eedge, emax_u, denom, E);
    edge_scatter<<<E, 256, 0, stream>>>(src, dst, bufA, eedge, denom, bufC, E);
    bias_elu<<<gNF, thr, 0, stream>>>(bufC, b2, NF, HID);

    // ---------------- final FC + elu ----------------
    pack_b<<<pack_blocks(64 >> 4, HID), thr, 0, stream>>>(Wfc, Bpk, HID, 64);
    pack_a<<<pack_blocks(N >> 4, HID), thr, 0, stream>>>(bufC, Apk2, N, HID);
    gemm_bf16_wmma<<<gemm_blocks(N, 64), thr, 0, stream>>>(Apk2, Bpk, out, N, 64, HID);
    long long nOut = (long long)N * 64;
    bias_elu<<<(int)((nOut + thr - 1) / thr), thr, 0, stream>>>(out, bfc, nOut, 64);
}
